// LinearMultiheadAttention_2860448219985
// MI455X (gfx1250) — matrix operand order repaired and checked by reading, not compiled, not run on hardware
//
#include <hip/hip_runtime.h>
#include <hip/hip_bf16.h>
#include <stdint.h>

// Problem constants (match reference)
#define D_MODEL 1024
#define NHEAD   16
#define HEAD    64
#define BATCH   2
#define SEQ     4096
#define ROWS    (BATCH * SEQ)          // 8192
#define EPSV    1e-5f

// GEMM register blocking: each wave computes a 64x32 output block = 4x2 WMMA tiles
#define BM 64
#define BN 32

typedef __bf16 bf16_t;
typedef __attribute__((ext_vector_type(16))) __bf16 v16bf;
typedef __attribute__((ext_vector_type(8)))  float  v8f;

union Frag16 {
    v16bf v;
    uint4 q[2];
};

// ---------------------------------------------------------------------------
// fp32 -> bf16 cast (RNE via compiler conversion)
// ---------------------------------------------------------------------------
__global__ void cast_bf16_kernel(const float* __restrict__ src,
                                 bf16_t* __restrict__ dst, int n) {
    int i = blockIdx.x * blockDim.x + threadIdx.x;
    if (i < n) dst[i] = (bf16_t)src[i];
}

// ---------------------------------------------------------------------------
// WMMA GEMM: Y = X @ W^T + bias,   X: [M x 1024] bf16 row-major,
//                                  W: [1024 x 1024] bf16 row-major (used as W^T)
// One wave computes a 64x32 block (4x2 tiles) with v_wmma_f32_16x16x32_bf16:
// per k-step, 6 fragment loads feed 8 WMMAs (1.5 b128 loads per WMMA).
// mode 0: out[m,n] = y                       (row-major f32, final projection)
// mode 1: out[b,h,l,e] = exp(y)              (Q, head-split layout)
// mode 2: out[b,h,l,e] = exp(y) * mask[b,l]  (K)
// mode 3: out[b,h,l,e] = y * mask[b,l]       (V)
// ---------------------------------------------------------------------------
__global__ __launch_bounds__(32) void gemm_bf16_wmma_kernel(
    const bf16_t* __restrict__ X, const bf16_t* __restrict__ W,
    const float* __restrict__ bias, float* __restrict__ out,
    const unsigned char* __restrict__ mask, int mode)
{
    const int lane = threadIdx.x;      // 0..31, wave32
    const int h    = lane >> 4;        // half-wave select
    const int ln16 = lane & 15;
    const int n0   = blockIdx.x * BN;
    const int m0   = blockIdx.y * BM;

    // A lane: row m; 16-bit A 16x32 layout: VGPR v<4 -> K=2v+8h,
    // v>=4 -> K=16+2(v-4)+8h  => two contiguous 8-element (16B) chunks.
    const bf16_t* rowA0 = X + (size_t)(m0 +  0 + ln16) * D_MODEL;
    const bf16_t* rowA1 = X + (size_t)(m0 + 16 + ln16) * D_MODEL;
    const bf16_t* rowA2 = X + (size_t)(m0 + 32 + ln16) * D_MODEL;
    const bf16_t* rowA3 = X + (size_t)(m0 + 48 + ln16) * D_MODEL;
    // B lane: col n; same K order as A: elems 0-7 K=8h..8h+7,
    // elems 8-15 K=16+8h..16+8h+7 => two contiguous 8-element (16B) chunks.
    const bf16_t* rowB0 = W + (size_t)(n0 +  0 + ln16) * D_MODEL;
    const bf16_t* rowB1 = W + (size_t)(n0 + 16 + ln16) * D_MODEL;

    v8f acc[4][2];
    #pragma unroll
    for (int i = 0; i < 4; ++i)
        #pragma unroll
        for (int j = 0; j < 2; ++j)
            acc[i][j] = (v8f){};

    const int aoff = 8 * h;    // A fragment K offset for this half-wave
    const int boff = 8 * h;    // B fragment K offset for this half-wave

    for (int k0 = 0; k0 < D_MODEL; k0 += 32) {
        __builtin_prefetch(rowA0 + k0 + 256, 0, 3);
        __builtin_prefetch(rowB0 + k0 + 256, 0, 3);
        Frag16 a0, a1, a2, a3, b0, b1;
        a0.q[0] = *(const uint4*)(rowA0 + k0 + aoff);
        a0.q[1] = *(const uint4*)(rowA0 + k0 + 16 + aoff);
        a1.q[0] = *(const uint4*)(rowA1 + k0 + aoff);
        a1.q[1] = *(const uint4*)(rowA1 + k0 + 16 + aoff);
        a2.q[0] = *(const uint4*)(rowA2 + k0 + aoff);
        a2.q[1] = *(const uint4*)(rowA2 + k0 + 16 + aoff);
        a3.q[0] = *(const uint4*)(rowA3 + k0 + aoff);
        a3.q[1] = *(const uint4*)(rowA3 + k0 + 16 + aoff);
        b0.q[0] = *(const uint4*)(rowB0 + k0 + boff);
        b0.q[1] = *(const uint4*)(rowB0 + k0 + 16 + boff);
        b1.q[0] = *(const uint4*)(rowB1 + k0 + boff);
        b1.q[1] = *(const uint4*)(rowB1 + k0 + 16 + boff);

        acc[0][0] = __builtin_amdgcn_wmma_f32_16x16x32_bf16(
            false, a0.v, false, b0.v, (short)0, acc[0][0], false, false);
        acc[0][1] = __builtin_amdgcn_wmma_f32_16x16x32_bf16(
            false, a0.v, false, b1.v, (short)0, acc[0][1], false, false);
        acc[1][0] = __builtin_amdgcn_wmma_f32_16x16x32_bf16(
            false, a1.v, false, b0.v, (short)0, acc[1][0], false, false);
        acc[1][1] = __builtin_amdgcn_wmma_f32_16x16x32_bf16(
            false, a1.v, false, b1.v, (short)0, acc[1][1], false, false);
        acc[2][0] = __builtin_amdgcn_wmma_f32_16x16x32_bf16(
            false, a2.v, false, b0.v, (short)0, acc[2][0], false, false);
        acc[2][1] = __builtin_amdgcn_wmma_f32_16x16x32_bf16(
            false, a2.v, false, b1.v, (short)0, acc[2][1], false, false);
        acc[3][0] = __builtin_amdgcn_wmma_f32_16x16x32_bf16(
            false, a3.v, false, b0.v, (short)0, acc[3][0], false, false);
        acc[3][1] = __builtin_amdgcn_wmma_f32_16x16x32_bf16(
            false, a3.v, false, b1.v, (short)0, acc[3][1], false, false);
    }

    // C/D layout: lane half h, VGPR r -> M = r + 8h (within tile), N = ln16
    #pragma unroll
    for (int i = 0; i < 4; ++i) {
        #pragma unroll
        for (int j = 0; j < 2; ++j) {
            #pragma unroll
            for (int r = 0; r < 8; ++r) {
                const int m = m0 + 16 * i + r + 8 * h;
                const int n = n0 + 16 * j + ln16;
                float y = acc[i][j][r] + bias[n];
                if (mode == 0) {
                    out[(size_t)m * D_MODEL + n] = y;
                } else {
                    const int b  = m >> 12;          // m / SEQ
                    const int l  = m & (SEQ - 1);
                    const int hd = n >> 6;           // n / HEAD
                    const int e  = n & (HEAD - 1);
                    if (mode == 1) {
                        y = __expf(y);
                    } else {
                        const float mk = mask[b * SEQ + l] ? 1.0f : 0.0f;
                        y = (mode == 2) ? __expf(y) * mk : y * mk;
                    }
                    out[(((size_t)(b * NHEAD + hd) * SEQ) + l) * HEAD + e] = y;
                }
            }
        }
    }
}

// ---------------------------------------------------------------------------
// Per (b,h): Ksum[d] = sum_l K[l,d];  KV[d,e] = sum_l K[l,d]*V[l,e]
// 64 threads; thread e keeps a 64-deep register column of KV.
// ---------------------------------------------------------------------------
__global__ __launch_bounds__(64) void kv_ksum_kernel(
    const float* __restrict__ K, const float* __restrict__ V,
    float* __restrict__ KsumOut, float* __restrict__ KVOut)
{
    const int bh = blockIdx.x;
    const int e  = threadIdx.x;
    const float* Kb = K + (size_t)bh * SEQ * HEAD;
    const float* Vb = V + (size_t)bh * SEQ * HEAD;

    __shared__ float kbuf[HEAD];
    float kv[HEAD];
    #pragma unroll
    for (int d = 0; d < HEAD; ++d) kv[d] = 0.0f;
    float ksum = 0.0f;

    for (int l = 0; l < SEQ; ++l) {
        const float kval = Kb[(size_t)l * HEAD + e];
        ksum += kval;
        kbuf[e] = kval;
        __syncthreads();
        const float vle = Vb[(size_t)l * HEAD + e];
        #pragma unroll
        for (int d = 0; d < HEAD; ++d) kv[d] += kbuf[d] * vle;
        __syncthreads();
    }

    KsumOut[bh * HEAD + e] = ksum;
    float* kvb = KVOut + (size_t)bh * HEAD * HEAD;
    #pragma unroll
    for (int d = 0; d < HEAD; ++d) kvb[d * HEAD + e] = kv[d];
}

// ---------------------------------------------------------------------------
// norm[row] = Q[row,:] . Ksum[bh,:] + EPS    (one wave per row, shuffle reduce)
// ---------------------------------------------------------------------------
__global__ __launch_bounds__(256) void norm_base_kernel(
    const float* __restrict__ Q, const float* __restrict__ Ksum,
    float* __restrict__ normOut)
{
    const int warp = threadIdx.x >> 5;
    const int lane = threadIdx.x & 31;
    const int row  = blockIdx.x * 8 + warp;      // [0, B*H*SEQ)
    const int bh   = row >> 12;                  // row / SEQ
    const float* qrow = Q + (size_t)row * HEAD;
    const float* ks   = Ksum + bh * HEAD;
    float d0 = qrow[lane] * ks[lane] + qrow[lane + 32] * ks[lane + 32];
    for (int off = 16; off > 0; off >>= 1) d0 += __shfl_down(d0, off);
    if (lane == 0) normOut[row] = d0 + EPSV;
}

// ---------------------------------------------------------------------------
// Per (b,h): scalar scans cumKf (inclusive fwd) / cumKb (exclusive bwd),
// then norm += Qf*cumKf + Qb*cumKb and invert in place.
// ---------------------------------------------------------------------------
__global__ __launch_bounds__(256) void scan_inv_kernel(
    const unsigned char* __restrict__ mask, const float* __restrict__ base_amp,
    float* __restrict__ norm)
{
    __shared__ float cKf[SEQ];
    __shared__ float cKb[SEQ];
    const int bh = blockIdx.x;
    const int b  = bh >> 4;
    const int h  = bh & (NHEAD - 1);
    const float amp = __expf(base_amp[h]);

    if (threadIdx.x == 0) {
        float run = 0.0f;
        for (int l = 0; l < SEQ; ++l) {
            const float kf = mask[b * SEQ + l] ? __expf(-amp * (float)l) : 0.0f;
            run += kf;                 // inclusive
            cKf[l] = run;
        }
    } else if (threadIdx.x == 1) {
        float run = 0.0f;
        for (int l = SEQ - 1; l >= 0; --l) {
            cKb[l] = run;              // exclusive (sum over j > l)
            run += mask[b * SEQ + l] ? __expf(amp * (float)l) : 0.0f;
        }
    }
    __syncthreads();

    float* nrm = norm + (size_t)bh * SEQ;
    for (int l = threadIdx.x; l < SEQ; l += 256) {
        const float qf = __expf(amp * (float)l);
        const float qb = __expf(-amp * (float)l);
        const float nv = nrm[l] + qf * cKf[l] + qb * cKb[l];
        nrm[l] = 1.0f / nv;
    }
}

// ---------------------------------------------------------------------------
// Per (b,h): O[l,e] = inv[l]*( Q[l,:].KV[:,e] + Qf[l]*cumKfV[l,e] )
//            then  += inv[l]*Qb[l]*cumKbV[l,e]  (backward pass)
// Running cum_KfV / cum_KbV live in a register (per-thread column e).
// ---------------------------------------------------------------------------
__global__ __launch_bounds__(64) void out_attn_kernel(
    const float* __restrict__ Q, const float* __restrict__ V,
    const float* __restrict__ KV, const float* __restrict__ inv,
    const unsigned char* __restrict__ mask, const float* __restrict__ base_amp,
    float* __restrict__ O)
{
    const int bh = blockIdx.x;
    const int b  = bh >> 4;
    const int h  = bh & (NHEAD - 1);
    const int e  = threadIdx.x;
    const float amp = __expf(base_amp[h]);

    const float* Qb   = Q   + (size_t)bh * SEQ * HEAD;
    const float* Vb   = V   + (size_t)bh * SEQ * HEAD;
    const float* invb = inv + (size_t)bh * SEQ;
    float*       Ob   = O   + (size_t)bh * SEQ * HEAD;

    float kv[HEAD];
    #pragma unroll
    for (int d = 0; d < HEAD; ++d)
        kv[d] = KV[(size_t)bh * HEAD * HEAD + d * HEAD + e];

    __shared__ float qbuf[HEAD];

    // forward pass: KV term + forward positional term (inclusive cumsum)
    float accF = 0.0f;
    for (int l = 0; l < SEQ; ++l) {
        qbuf[e] = Qb[(size_t)l * HEAD + e];
        __syncthreads();
        const float mk = mask[b * SEQ + l] ? 1.0f : 0.0f;
        accF += mk * __expf(-amp * (float)l) * Vb[(size_t)l * HEAD + e];
        float dot = 0.0f;
        #pragma unroll
        for (int d = 0; d < HEAD; ++d) dot += qbuf[d] * kv[d];
        const float invl = invb[l];
        Ob[(size_t)l * HEAD + e] = invl * (dot + __expf(amp * (float)l) * accF);
        __syncthreads();
    }

    // backward pass: exclusive suffix sum term
    float accB = 0.0f;
    for (int l = SEQ - 1; l >= 0; --l) {
        const float vle  = Vb[(size_t)l * HEAD + e];
        const float invl = invb[l];
        Ob[(size_t)l * HEAD + e] += invl * __expf(-amp * (float)l) * accB;
        const float mk = mask[b * SEQ + l] ? 1.0f : 0.0f;
        accB += mk * __expf(amp * (float)l) * vle;
    }
}

// ---------------------------------------------------------------------------
// Join heads [b,h,l,e] -> [b,l,h*64+e] and cast to bf16 for final GEMM
// ---------------------------------------------------------------------------
__global__ void join_cast_kernel(const float* __restrict__ O,
                                 bf16_t* __restrict__ dst) {
    const int idx = blockIdx.x * blockDim.x + threadIdx.x;
    if (idx >= BATCH * NHEAD * SEQ * HEAD) return;
    const int e = idx & (HEAD - 1);
    const int l = (idx >> 6) & (SEQ - 1);
    const int h = (idx >> 18) & (NHEAD - 1);
    const int b = idx >> 22;
    dst[((size_t)(b * SEQ + l)) * D_MODEL + h * HEAD + e] = (bf16_t)O[idx];
}

// ---------------------------------------------------------------------------
// Host-side launcher
// ---------------------------------------------------------------------------
extern "C" void kernel_launch(void* const* d_in, const int* in_sizes, int n_in,
                              void* d_out, int out_size, void* d_ws, size_t ws_size,
                              hipStream_t stream) {
    const float*         queries  = (const float*)d_in[0];
    const unsigned char* mask     = (const unsigned char*)d_in[1];   // jax bool = 1B
    const float*         Wq       = (const float*)d_in[2];
    const float*         bq       = (const float*)d_in[3];
    const float*         Wk       = (const float*)d_in[4];
    const float*         bk       = (const float*)d_in[5];
    const float*         Wv       = (const float*)d_in[6];
    const float*         bv       = (const float*)d_in[7];
    const float*         Wo       = (const float*)d_in[8];
    const float*         bo       = (const float*)d_in[9];
    const float*         base_amp = (const float*)d_in[10];

    char* ws = (char*)d_ws;
    size_t off = 0;
    auto alloc = [&](size_t bytes) -> void* {
        void* p = ws + off;
        off += (bytes + 255) & ~(size_t)255;
        return p;
    };

    bf16_t* Xbf  = (bf16_t*)alloc((size_t)ROWS * D_MODEL * sizeof(bf16_t));
    bf16_t* Wqb  = (bf16_t*)alloc((size_t)D_MODEL * D_MODEL * sizeof(bf16_t));
    bf16_t* Wkb  = (bf16_t*)alloc((size_t)D_MODEL * D_MODEL * sizeof(bf16_t));
    bf16_t* Wvb  = (bf16_t*)alloc((size_t)D_MODEL * D_MODEL * sizeof(bf16_t));
    bf16_t* Wob  = (bf16_t*)alloc((size_t)D_MODEL * D_MODEL * sizeof(bf16_t));
    float*  Qh   = (float*)alloc((size_t)BATCH * NHEAD * SEQ * HEAD * sizeof(float));
    float*  Kh   = (float*)alloc((size_t)BATCH * NHEAD * SEQ * HEAD * sizeof(float));
    float*  Vh   = (float*)alloc((size_t)BATCH * NHEAD * SEQ * HEAD * sizeof(float));
    float*  Oh   = (float*)alloc((size_t)BATCH * NHEAD * SEQ * HEAD * sizeof(float));
    bf16_t* Obf  = (bf16_t*)alloc((size_t)ROWS * D_MODEL * sizeof(bf16_t));
    float*  Ksum = (float*)alloc((size_t)BATCH * NHEAD * HEAD * sizeof(float));
    float*  KV   = (float*)alloc((size_t)BATCH * NHEAD * HEAD * HEAD * sizeof(float));
    float*  nrm  = (float*)alloc((size_t)BATCH * NHEAD * SEQ * sizeof(float));
    (void)ws_size; (void)in_sizes; (void)n_in; (void)out_size;

    // 1) fp32 -> bf16 casts
    const int nX = ROWS * D_MODEL;
    const int nW = D_MODEL * D_MODEL;
    cast_bf16_kernel<<<(nX + 255) / 256, 256, 0, stream>>>(queries, Xbf, nX);
    cast_bf16_kernel<<<(nW + 255) / 256, 256, 0, stream>>>(Wq, Wqb, nW);
    cast_bf16_kernel<<<(nW + 255) / 256, 256, 0, stream>>>(Wk, Wkb, nW);
    cast_bf16_kernel<<<(nW + 255) / 256, 256, 0, stream>>>(Wv, Wvb, nW);
    cast_bf16_kernel<<<(nW + 255) / 256, 256, 0, stream>>>(Wo, Wob, nW);

    // 2) QKV projections via WMMA (fused epilogue: bias, exp, mask, head split)
    dim3 ggrid(D_MODEL / BN, ROWS / BM);
    gemm_bf16_wmma_kernel<<<ggrid, 32, 0, stream>>>(Xbf, Wqb, bq, Qh, mask, 1);
    gemm_bf16_wmma_kernel<<<ggrid, 32, 0, stream>>>(Xbf, Wkb, bk, Kh, mask, 2);
    gemm_bf16_wmma_kernel<<<ggrid, 32, 0, stream>>>(Xbf, Wvb, bv, Vh, mask, 3);

    // 3) linear-attention state + normalization + O
    kv_ksum_kernel<<<BATCH * NHEAD, 64, 0, stream>>>(Kh, Vh, Ksum, KV);
    norm_base_kernel<<<BATCH * NHEAD * SEQ / 8, 256, 0, stream>>>(Qh, Ksum, nrm);
    scan_inv_kernel<<<BATCH * NHEAD, 256, 0, stream>>>(mask, base_amp, nrm);
    out_attn_kernel<<<BATCH * NHEAD, 64, 0, stream>>>(Qh, Vh, KV, nrm, mask,
                                                      base_amp, Oh);

    // 4) join heads, cast, output projection via WMMA
    const int nO = BATCH * NHEAD * SEQ * HEAD;
    join_cast_kernel<<<(nO + 255) / 256, 256, 0, stream>>>(Oh, Obf);
    gemm_bf16_wmma_kernel<<<ggrid, 32, 0, stream>>>(Obf, Wob, bo, (float*)d_out,
                                                    mask, 0);
}
